// GraphSAGE_DGL_32289564131886
// MI455X (gfx1250) — matrix operand order repaired and checked by reading, not compiled, not run on hardware
//
#include <hip/hip_runtime.h>
#include <stdint.h>

// ---- vector types for WMMA fragments ----
typedef __attribute__((ext_vector_type(16))) __bf16 v16bf;
typedef __attribute__((ext_vector_type(8)))  __bf16 v8bf;
typedef __attribute__((ext_vector_type(8)))  float  v8f;

__device__ __forceinline__ __bf16 f2bf(float f) { return (__bf16)f; }

// Deterministic hash-based dropout (p = 0.5, inverted scale 2.0).
__device__ __forceinline__ float drop_mask(uint32_t m, uint32_t c, uint32_t layer) {
  uint32_t x = m * 2654435761u ^ (c * 0x9E3779B9u) ^ (layer * 0x85EBCA6Bu + 0xC2B2AE35u);
  x ^= x >> 16; x *= 0x7FEB352Du; x ^= x >> 15; x *= 0x846CA68Bu; x ^= x >> 16;
  return (x & 1u) ? 2.0f : 0.0f;
}

// -------- edge scatter: agg[dst] += h[src], cnt[dst] += 1 --------
// one block per edge, one lane per feature; atomics land in L2 (accumulator << 192MB L2)
__global__ void sage_scatter(const float* __restrict__ h,
                             const int* __restrict__ src,
                             const int* __restrict__ dst,
                             float* __restrict__ agg,
                             float* __restrict__ cnt, int d) {
  int e = blockIdx.x;
  int s = src[e];
  int t = dst[e];
  int k = threadIdx.x;
  atomicAdd(&agg[(size_t)t * d + k], h[(size_t)s * d + k]);
  if (k == 0) atomicAdd(&cnt[t], 1.0f);
}

// -------- fused mean + f32->bf16 convert of both GEMM A operands --------
__global__ void sage_prep_bf16(const float* __restrict__ hin,
                               const float* __restrict__ agg,
                               const float* __restrict__ cnt,
                               __bf16* __restrict__ selfA,
                               __bf16* __restrict__ neighA,
                               int kShift, long long total) {
  long long t = (long long)blockIdx.x * blockDim.x + threadIdx.x;
  if (t >= total) return;
  long long i = t >> kShift;                 // row (K is a power of two)
  float c = fmaxf(cnt[i], 1.0f);
  selfA[t]  = f2bf(hin[t]);
  neighA[t] = f2bf(agg[t] / c);
}

// -------- weight transpose to column-major bf16, zero-padded to Nc columns --------
// WT[n*K + k] = (n < N) ? W[k*N + n] : 0   for n in [0, Nc)
__global__ void sage_wT_bf16(const float* __restrict__ Ws,
                             const float* __restrict__ Wn,
                             __bf16* __restrict__ WsT,
                             __bf16* __restrict__ WnT, int K, int N, int Nc) {
  int t = blockIdx.x * blockDim.x + threadIdx.x;
  if (t >= K * Nc) return;
  int n = t / K, k = t - n * K;
  float vs = 0.0f, vn = 0.0f;
  if (n < N) {
    vs = Ws[(size_t)k * N + n];
    vn = Wn[(size_t)k * N + n];
  }
  WsT[t] = f2bf(vs);
  WnT[t] = f2bf(vn);
}

// -------- WMMA GEMM: out = selfA@Ws + neighA@Wn + b  (+relu +dropout) --------
// One wave computes a 16x64 output strip (4 adjacent 16x16 N-tiles). The A
// fragments (self + neigh) are loaded once per K-step and reused across 8
// v_wmma_f32_16x16x32_bf16 ops. The 4 tiles' B columns live at a fixed
// 16*K-element stride in the padded transposed weights, so all B loads share
// one base address per matrix with pure immediate offsets (frees VGPRs for
// load/WMMA pipelining instead of 64-bit address arithmetic).
#define NT 4
__global__ void __launch_bounds__(256)
sage_gemm_wmma(const __bf16* __restrict__ selfA,
               const __bf16* __restrict__ neighA,
               const __bf16* __restrict__ WsT,
               const __bf16* __restrict__ WnT,
               const float* __restrict__ bias,
               float* __restrict__ out,
               int M, int K, int N, int nGroups, int totalWaves,
               int reluDrop, int layer) {
  int wid = (int)((blockIdx.x * (unsigned)blockDim.x + threadIdx.x) >> 5);
  if (wid >= totalWaves) return;             // wave-uniform exit
  int lane = threadIdx.x & 31;
  int half = lane >> 4;                      // lane group 0-15 / 16-31
  int n    = lane & 15;
  int mTile = wid / nGroups;
  int g     = wid - mTile * nGroups;
  int mBase = mTile << 4;
  int colBase = (g * NT) << 4;               // first column of this wave's strip

  // A rows clamped (a D row depends only on the matching A row; clamped rows
  // are never stored). EXEC stays all-ones through every WMMA.
  int am  = mBase + n;
  int amC = am < M ? am : (M - 1);
  const __bf16* aS = selfA  + (size_t)amC * K;
  const __bf16* aN = neighA + (size_t)amC * K;

  // Single B base per matrix; tiles t=0..3 live at immediate offset t*16*K.
  const __bf16* bS = WsT + (size_t)(colBase + n) * K;
  const __bf16* bN = WnT + (size_t)(colBase + n) * K;
  const int tileStride = K << 4;             // 16*K elements between N-tiles

  v8f acc[NT];
#pragma unroll
  for (int t = 0; t < NT; ++t) acc[t] = (v8f){};

  for (int k0 = 0; k0 < K; k0 += 32) {
    // A fragment (contiguous k order, same as B): half 0 -> K {0..15}, half 1 -> {16..31}
    int ka0 = k0 + half * 16;
    int ka1 = k0 + half * 16 + 8;
    v8bf s0 = *(const v8bf*)(aS + ka0);
    v8bf s1 = *(const v8bf*)(aS + ka1);
    v8bf q0 = *(const v8bf*)(aN + ka0);
    v8bf q1 = *(const v8bf*)(aN + ka1);
    v16bf A0, A1;
#pragma unroll
    for (int j = 0; j < 8; ++j) {
      A0[j] = s0[j]; A0[8 + j] = s1[j];
      A1[j] = q0[j]; A1[8 + j] = q1[j];
    }
    // B fragments: 16 contiguous K values of this lane's column per tile.
    int kb = k0 + half * 16;
    v16bf B0[NT], B1[NT];
#pragma unroll
    for (int t = 0; t < NT; ++t) {
      B0[t] = *(const v16bf*)(bS + t * tileStride + kb);
      B1[t] = *(const v16bf*)(bN + t * tileStride + kb);
    }
#pragma unroll
    for (int t = 0; t < NT; ++t) {
      acc[t] = __builtin_amdgcn_wmma_f32_16x16x32_bf16(false, A0, false, B0[t],
                                                       (short)0, acc[t], false, false);
      acc[t] = __builtin_amdgcn_wmma_f32_16x16x32_bf16(false, A1, false, B1[t],
                                                       (short)0, acc[t], false, false);
    }
  }

#pragma unroll
  for (int t = 0; t < NT; ++t) {
    int col = colBase + (t << 4) + n;
    if (col < N) {
      float bv = bias[col];
#pragma unroll
      for (int r = 0; r < 8; ++r) {
        int m = mBase + half * 8 + r;        // C/D layout: VGPR r -> M = r + 8*half
        if (m < M) {
          float v = acc[t][r] + bv;
          if (reluDrop) {
            v = fmaxf(v, 0.0f);
            v *= drop_mask((uint32_t)m, (uint32_t)col, (uint32_t)layer);
          }
          out[(size_t)m * N + col] = v;
        }
      }
    }
  }
}

extern "C" void kernel_launch(void* const* d_in, const int* in_sizes, int n_in,
                              void* d_out, int out_size, void* d_ws, size_t ws_size,
                              hipStream_t stream) {
  const float* h   = (const float*)d_in[0];
  const float* W0s = (const float*)d_in[1];
  const float* W0n = (const float*)d_in[2];
  const float* b0  = (const float*)d_in[3];
  const float* W1s = (const float*)d_in[4];
  const float* W1n = (const float*)d_in[5];
  const float* b1  = (const float*)d_in[6];
  const float* W2s = (const float*)d_in[7];
  const float* W2n = (const float*)d_in[8];
  const float* b2  = (const float*)d_in[9];
  const int* e0s = (const int*)d_in[10];
  const int* e0d = (const int*)d_in[11];
  const int* e1s = (const int*)d_in[12];
  const int* e1d = (const int*)d_in[13];
  const int* e2s = (const int*)d_in[14];
  const int* e2d = (const int*)d_in[15];
  float* out = (float*)d_out;

  auto al = [](size_t x) { return (x + 255) & ~(size_t)255; };
  char* w = (char*)d_ws;
  float*  h1     = (float*)w;  w += al((size_t)50000 * 256 * 4);   // layer0 out
  float*  h2     = (float*)w;  w += al((size_t)12500 * 256 * 4);   // layer1 out
  float*  agg    = (float*)w;  w += al((size_t)50000 * 128 * 4);   // max over layers
  float*  cnt    = (float*)w;  w += al((size_t)50000 * 4);
  __bf16* selfA  = (__bf16*)w; w += al((size_t)50000 * 128 * 2);   // max elems
  __bf16* neighA = (__bf16*)w; w += al((size_t)50000 * 128 * 2);
  __bf16* WsT    = (__bf16*)w; w += al((size_t)256 * 256 * 2);     // K*Nc max
  __bf16* WnT    = (__bf16*)w; w += al((size_t)256 * 256 * 2);

  auto runLayer = [&](const float* hin, const int* es, const int* ed, int E,
                      int Mdst, int K, int N,
                      const float* Ws, const float* Wn, const float* bias,
                      float* hout, int reluDrop, int layer) {
    hipMemsetAsync(agg, 0, (size_t)Mdst * K * sizeof(float), stream);
    hipMemsetAsync(cnt, 0, (size_t)Mdst * sizeof(float), stream);
    sage_scatter<<<E, K, 0, stream>>>(hin, es, ed, agg, cnt, K);

    long long total = (long long)Mdst * K;
    int kShift = (K == 128) ? 7 : 8;
    int pBlocks = (int)((total + 255) / 256);
    sage_prep_bf16<<<pBlocks, 256, 0, stream>>>(hin, agg, cnt, selfA, neighA,
                                                kShift, total);

    int Nc = ((N + NT * 16 - 1) / (NT * 16)) * (NT * 16);  // pad cols to 64
    int wtot = K * Nc;
    sage_wT_bf16<<<(wtot + 255) / 256, 256, 0, stream>>>(Ws, Wn, WsT, WnT,
                                                         K, N, Nc);

    int nGroups = Nc / (NT * 16);            // 4 N-tiles per wave
    int mTiles  = (Mdst + 15) / 16;
    int totalWaves = nGroups * mTiles;
    long long gThreads = (long long)totalWaves * 32;
    int gBlocks = (int)((gThreads + 255) / 256);
    sage_gemm_wmma<<<gBlocks, 256, 0, stream>>>(selfA, neighA, WsT, WnT, bias,
                                                hout, Mdst, K, N, nGroups,
                                                totalWaves, reluDrop, layer);
  };

  // layer 0: h[200000,128] -> h1[50000,256]
  runLayer(h,  e0s, e0d, 800000, 50000, 128, 256, W0s, W0n, b0, h1, 1, 0);
  // layer 1: h1 -> h2[12500,256]
  runLayer(h1, e1s, e1d, 200000, 12500, 256, 256, W1s, W1n, b1, h2, 1, 1);
  // layer 2: h2 -> out[3125,47]
  runLayer(h2, e2s, e2d,  50000,  3125, 256,  47, W2s, W2n, b2, out, 0, 2);
}
